// GATLayer_32564442038893
// MI455X (gfx1250) — matrix operand order repaired and checked by reading, not compiled, not run on hardware
//
#include <hip/hip_runtime.h>
#include <hip/hip_fp16.h>

typedef __attribute__((ext_vector_type(16))) _Float16 v16h;
typedef __attribute__((ext_vector_type(8)))  float    v8f;

#define B_    8
#define N_    2048
#define F_    64
#define ALPHA 0.2f
#define NEGV  -9.0e15f

// ---------------------------------------------------------------------------
// Kernel 1: Wh = h @ W^T (fp32 VALU, exact for the logits), stored transposed
// in f16 as WhT[b][o][n] so the WMMA B-operand is a contiguous 32B load.
// Also e1 = Wh@a[:64], e2 = Wh@a[64:].
// One 64-thread block per (b, n) row; thread t computes output feature t.
// ---------------------------------------------------------------------------
__global__ void __launch_bounds__(64) gat_prep(
    const float* __restrict__ h, const float* __restrict__ W,
    const float* __restrict__ a, _Float16* __restrict__ WhT,
    float* __restrict__ e1, float* __restrict__ e2) {
  const int row = blockIdx.x;          // 0 .. B*N-1
  const int b   = row >> 11;           // row / N_
  const int n   = row & (N_ - 1);
  const int t   = threadIdx.x;         // 0..63 (output feature o)

  __shared__ float hrow[F_];
  __shared__ float red[F_];

  hrow[t] = h[(size_t)row * F_ + t];
  __syncthreads();

  float acc = 0.f;
#pragma unroll
  for (int k = 0; k < F_; ++k)
    acc = fmaf(hrow[k], W[t * F_ + k], acc);

  WhT[((size_t)b * F_ + t) * N_ + n] = (_Float16)acc;

  // e1[row] = sum_o Wh[row][o] * a[o]
  red[t] = acc * a[t];
  __syncthreads();
  for (int s = 32; s > 0; s >>= 1) {
    if (t < s) red[t] += red[t + s];
    __syncthreads();
  }
  if (t == 0) e1[row] = red[0];
  __syncthreads();

  // e2[row] = sum_o Wh[row][o] * a[64 + o]
  red[t] = acc * a[F_ + t];
  __syncthreads();
  for (int s = 32; s > 0; s >>= 1) {
    if (t < s) red[t] += red[t + s];
    __syncthreads();
  }
  if (t == 0) e2[row] = red[0];
}

// ---------------------------------------------------------------------------
// Kernel 2: fused masked-softmax attention + P@Wh via v_wmma_f32_16x16x32_f16.
// Grid (N/16, B); 128 threads = 4 wave32 per 16-query tile.
//   Phase 1: row max of masked leaky-relu logits (reference NEG semantics).
//   Phase 2: waves split the 64 j-chunks of K=32; each chunk builds the f16
//            P tile in contiguous k order (k = 16*half + e, as B) and does 4 WMMAs (16x64 output).
//   Epilogue: LDS reduction across waves, divide by row sum, ELU, store.
// ---------------------------------------------------------------------------
__global__ void __launch_bounds__(128) gat_attn(
    const float* __restrict__ adj, const _Float16* __restrict__ WhT,
    const float* __restrict__ e1, const float* __restrict__ e2,
    float* __restrict__ out) {
  const int b  = blockIdx.y;
  const int i0 = blockIdx.x * 16;
  const int t  = threadIdx.x;

  __shared__ float e1s[16];
  __shared__ float rowmax[16];
  __shared__ float rowsumS[16];
  __shared__ float part[16][8];
  __shared__ float accW[4][16][64];   // per-wave partial outputs (16 KB)
  __shared__ float rsW[4][16];        // per-wave partial row sums

  const float* e2b = e2 + (size_t)b * N_;
  if (t < 16) e1s[t] = e1[(size_t)b * N_ + i0 + t];
  __syncthreads();

  // ---- Phase 1: per-row max of (adj>0 ? leakyrelu(e1+e2) : NEG) ----
  {
    const int r = t >> 3;
    const int c = t & 7;
    const float ev = e1s[r];
    const float* arow = adj + (size_t)(i0 + r) * N_;
    float mx = -3.4e38f;
    for (int j = c; j < N_; j += 8) {
      float s  = ev + e2b[j];
      float lr = s > 0.f ? s : ALPHA * s;
      float v  = arow[j] > 0.f ? lr : NEGV;
      mx = fmaxf(mx, v);
    }
    part[r][c] = mx;
  }
  __syncthreads();
  if (t < 16) {
    float mx = part[t][0];
#pragma unroll
    for (int c = 1; c < 8; ++c) mx = fmaxf(mx, part[t][c]);
    rowmax[t] = mx;
  }
  __syncthreads();

  // ---- Phase 2: exp + sum + P@Wh, waves own disjoint j-chunks ----
  const int w    = t >> 5;   // wave 0..3
  const int l    = t & 31;   // lane
  const int m    = l & 15;   // A row / B col within tile
  const int half = l >> 4;

  const float ev   = e1s[m];
  const float rmax = rowmax[m];
  const float* arow = adj + (size_t)(i0 + m) * N_;
  const _Float16* wbase = WhT + (size_t)b * F_ * N_;

  v8f acc0 = {}, acc1 = {}, acc2 = {}, acc3 = {};
  float psum = 0.f;

  for (int it = 0; it < 16; ++it) {
    const int j0 = (it * 4 + w) * 32;
    // A operand: 16x32 f16, K = half*16 + e (matches B)
    v16h A;
#pragma unroll
    for (int e = 0; e < 16; ++e) {
      const int k = (half << 4) + e;
      const int j = j0 + k;
      float s  = ev + e2b[j];
      float lr = s > 0.f ? s : ALPHA * s;
      float v  = arow[j] > 0.f ? lr : NEGV;
      float p  = __expf(v - rmax);   // masked -> 0; all-masked row -> 1
      A[e] = (_Float16)p;
      psum += p;
    }
    // B operands: 32x16 f16, lane = col n=m, K = half*16 + e  (contiguous 32B)
    const int jb = j0 + half * 16;
    v16h B0 = *(const v16h*)(wbase + (size_t)( 0 + m) * N_ + jb);
    v16h B1 = *(const v16h*)(wbase + (size_t)(16 + m) * N_ + jb);
    v16h B2 = *(const v16h*)(wbase + (size_t)(32 + m) * N_ + jb);
    v16h B3 = *(const v16h*)(wbase + (size_t)(48 + m) * N_ + jb);
    acc0 = __builtin_amdgcn_wmma_f32_16x16x32_f16(false, A, false, B0,
                                                  (short)0, acc0, false, false);
    acc1 = __builtin_amdgcn_wmma_f32_16x16x32_f16(false, A, false, B1,
                                                  (short)0, acc1, false, false);
    acc2 = __builtin_amdgcn_wmma_f32_16x16x32_f16(false, A, false, B2,
                                                  (short)0, acc2, false, false);
    acc3 = __builtin_amdgcn_wmma_f32_16x16x32_f16(false, A, false, B3,
                                                  (short)0, acc3, false, false);
  }

  // lanes (l, l^16) together cover all 32 k of row m in each chunk
  float other = __shfl_xor(psum, 16, 32);
  if (half == 0) rsW[w][m] = psum + other;

  // C/D layout: VGPR r -> row (half ? r+8 : r), col = m
#pragma unroll
  for (int r = 0; r < 8; ++r) {
    const int row = half * 8 + r;
    accW[w][row][ 0 + m] = acc0[r];
    accW[w][row][16 + m] = acc1[r];
    accW[w][row][32 + m] = acc2[r];
    accW[w][row][48 + m] = acc3[r];
  }
  __syncthreads();

  if (t < 16)
    rowsumS[t] = rsW[0][t] + rsW[1][t] + rsW[2][t] + rsW[3][t];
  __syncthreads();

  // ---- Epilogue: reduce waves, divide, ELU, coalesced store ----
#pragma unroll
  for (int q = 0; q < 8; ++q) {
    const int idx = q * 128 + t;
    const int row = idx >> 6;
    const int col = idx & 63;
    float v = (accW[0][row][col] + accW[1][row][col] +
               accW[2][row][col] + accW[3][row][col]) / rowsumS[row];
    v = v > 0.f ? v : (__expf(v) - 1.f);   // elu, alpha=1
    out[((size_t)(b * N_ + i0 + row)) * F_ + col] = v;
  }
}

// ---------------------------------------------------------------------------
extern "C" void kernel_launch(void* const* d_in, const int* in_sizes, int n_in,
                              void* d_out, int out_size, void* d_ws, size_t ws_size,
                              hipStream_t stream) {
  (void)in_sizes; (void)n_in; (void)out_size; (void)ws_size;
  const float* h   = (const float*)d_in[0];
  const float* adj = (const float*)d_in[1];
  const float* W   = (const float*)d_in[2];
  const float* a   = (const float*)d_in[3];
  float* out = (float*)d_out;

  char* ws = (char*)d_ws;
  _Float16* WhT = (_Float16*)ws;                                  // 2 MB
  float* e1 = (float*)(ws + (size_t)B_ * F_ * N_ * sizeof(_Float16));
  float* e2 = e1 + (size_t)B_ * N_;

  gat_prep<<<B_ * N_, 64, 0, stream>>>(h, W, a, WhT, e1, e2);
  gat_attn<<<dim3(N_ / 16, B_), 128, 0, stream>>>(adj, WhT, e1, e2, out);
}
